// MultiScaleAttention_83485574299670
// MI455X (gfx1250) — matrix operand order repaired and checked by reading, not compiled, not run on hardware
//
#include <hip/hip_runtime.h>
#include <hip/hip_bf16.h>

// ---------------------------------------------------------------------------
// Types / helpers
// ---------------------------------------------------------------------------
typedef __attribute__((ext_vector_type(16))) _Float16 v16h;
typedef __attribute__((ext_vector_type(8)))  float    v8f;

struct alignas(16) U128 { unsigned x, y, z, w; };

union F16Frag {
    v16h     v;
    _Float16 h[16];
    unsigned u[8];
};

// A-operand fragment (16x32 f16, M x K) from an LDS tile stored row-major
// [row][k] with row-stride `ldh` halves. `tile` points at tile[rowbase][k0].
// ISA 7.12.2: lanes 0-15 -> K 0..7 & 16..23 ; lanes 16-31 -> K 8..15 & 24..31.
__device__ inline v16h lds_a_frag(const _Float16* tile, int ldh, int lane) {
    F16Frag f;
    const int m  = lane & 15;
    const int kb = (lane < 16) ? 0 : 4;               // in u32 units (8 halves)
    const unsigned* row = reinterpret_cast<const unsigned*>(tile + m * ldh);
#pragma unroll
    for (int j = 0; j < 4; ++j) f.u[j]     = row[kb + j];
#pragma unroll
    for (int j = 0; j < 4; ++j) f.u[4 + j] = row[8 + kb + j];
    return f.v;
}

// B-operand fragment (32x16 f16, K x N) from an LDS tile stored [n][k]
// (k contiguous). Lanes 0-15: K 0..7 & 16..23, lanes 16-31: K 8..15 & 24..31, N = lane & 15.
__device__ inline v16h lds_b_frag(const _Float16* tile, int ldh, int lane) {
    F16Frag f;
    const int n  = lane & 15;
    const int kb = (lane < 16) ? 0 : 8;               // halves
    const _Float16* p = tile + n * ldh + kb;
    *reinterpret_cast<U128*>(&f.u[0]) = *reinterpret_cast<const U128*>(p);
    *reinterpret_cast<U128*>(&f.u[4]) = *reinterpret_cast<const U128*>(p + 16);
    return f.v;
}

__device__ inline v8f wmma_f16(v16h a, v16h b, v8f c) {
    return __builtin_amdgcn_wmma_f32_16x16x32_f16(false, a, false, b,
                                                  (short)0, c, false, false);
}

__device__ inline float wave_red_sum(float v) {
#pragma unroll
    for (int off = 16; off > 0; off >>= 1) v += __shfl_down(v, off, 32);
    return v;
}

// ---------------------------------------------------------------------------
// fp32 -> fp16 convert, 8 elements / thread (b128 in, b128 out)
// ---------------------------------------------------------------------------
__global__ void cvt_f32_f16(const float* __restrict__ in,
                            _Float16* __restrict__ out, int n8) {
    for (int i = blockIdx.x * blockDim.x + threadIdx.x; i < n8;
         i += gridDim.x * blockDim.x) {
        const float* p = in + (size_t)i * 8;
        _Float16 h[8];
#pragma unroll
        for (int j = 0; j < 8; ++j) h[j] = (_Float16)p[j];
        *reinterpret_cast<U128*>(out + (size_t)i * 8) =
            *reinterpret_cast<const U128*>(h);
    }
}

// ---------------------------------------------------------------------------
// Tiled WMMA GEMM:  C[M,N] = A[M,K] * W[N,K]^T + bias
// 256 threads (8 waves), 128x128 tile, 64-wide K chunks (16 WMMA/wave/stage),
// double-buffered dynamic LDS (one barrier per stage), register prefetch of
// the next chunk, L2 prefetch (global_prefetch_b8) of the chunk after that.
// ---------------------------------------------------------------------------
#define GEMM_LD   72                     // 64 + 8 pad halves (144B rows)
#define GEMM_TILE (128 * GEMM_LD)        // halves per tile
#define GEMM_SMEM (4 * GEMM_TILE * 2)    // bytes: 2 tiles x 2 buffers

extern __shared__ _Float16 dyn_smem[];

template <bool OUTF, bool OUTH, bool RELU>
__global__ __launch_bounds__(256) void gemm_wmma(
    const _Float16* __restrict__ A,
    const _Float16* __restrict__ W,
    const float*    __restrict__ bias,
    float*          __restrict__ Cf,
    _Float16*       __restrict__ Ch,
    int M, int N, int K)
{
    const int t    = threadIdx.x;
    const int lane = t & 31;
    const int wave = t >> 5;
    const int wm   = (wave >> 1) * 32;
    const int wn   = (wave & 1) * 64;
    const int rowbase = blockIdx.y * 128;
    const int colbase = blockIdx.x * 128;

    v8f acc[2][4] = {};
    U128 pa[4], pb[4];

    // 128 rows x 64 halves per tile = 1024 x 16B; 4 per thread per tile.
    auto gload = [&](int kt) {
        const int k0 = kt << 6;
#pragma unroll
        for (int i = 0; i < 4; ++i) {
            int idx = t + i * 256;
            int r = idx >> 3, s = (idx & 7) << 3;
            pa[i] = *reinterpret_cast<const U128*>(&A[(size_t)(rowbase + r) * K + k0 + s]);
            pb[i] = *reinterpret_cast<const U128*>(&W[(size_t)(colbase + r) * K + k0 + s]);
        }
    };
    auto sstore = [&](int buf) {
        _Float16* As = dyn_smem + buf * (2 * GEMM_TILE);
        _Float16* Bs = As + GEMM_TILE;
#pragma unroll
        for (int i = 0; i < 4; ++i) {
            int idx = t + i * 256;
            int r = idx >> 3, s = (idx & 7) << 3;
            *reinterpret_cast<U128*>(&As[r * GEMM_LD + s]) = pa[i];
            *reinterpret_cast<U128*>(&Bs[r * GEMM_LD + s]) = pb[i];
        }
    };

    gload(0);
    sstore(0);
    __syncthreads();

    const int nk = K >> 6;
    for (int kt = 0; kt < nk; ++kt) {
        if (kt + 1 < nk) gload(kt + 1);          // VGPR prefetch, next stage
        if (kt + 2 < nk) {                        // L2 prefetch, stage after
            const int k2 = (kt + 2) << 6;
            const _Float16* pf = (t < 128)
                ? &A[(size_t)(rowbase + t) * K + k2]
                : &W[(size_t)(colbase + (t - 128)) * K + k2];
            __builtin_prefetch(pf, 0, 3);
        }
        const _Float16* As = dyn_smem + (kt & 1) * (2 * GEMM_TILE);
        const _Float16* Bs = As + GEMM_TILE;
#pragma unroll
        for (int ks = 0; ks < 2; ++ks) {
            v16h af[2], bf[4];
#pragma unroll
            for (int g = 0; g < 2; ++g)
                af[g] = lds_a_frag(&As[(wm + g * 16) * GEMM_LD + ks * 32], GEMM_LD, lane);
#pragma unroll
            for (int c = 0; c < 4; ++c)
                bf[c] = lds_b_frag(&Bs[(wn + c * 16) * GEMM_LD + ks * 32], GEMM_LD, lane);
#pragma unroll
            for (int g = 0; g < 2; ++g)
#pragma unroll
                for (int c = 0; c < 4; ++c)
                    acc[g][c] = wmma_f16(af[g], bf[c], acc[g][c]);
        }
        if (kt + 1 < nk) sstore((kt + 1) & 1);   // fill other buffer, overlaps compute
        __syncthreads();
    }

    const int mo = (lane < 16) ? 0 : 8;
    const int no = lane & 15;
#pragma unroll
    for (int c = 0; c < 4; ++c) {
        const int col = colbase + wn + c * 16 + no;
        const float bv = bias[col];
#pragma unroll
        for (int g = 0; g < 2; ++g) {
            const size_t rb = (size_t)(rowbase + wm + g * 16 + mo) * N + col;
#pragma unroll
            for (int r = 0; r < 8; ++r) {
                float v = acc[g][c][r] + bv;
                if (RELU) v = fmaxf(v, 0.0f);
                if (OUTF) Cf[rb + (size_t)r * N] = v;
                if (OUTH) Ch[rb + (size_t)r * N] = (_Float16)v;
            }
        }
    }
}

// ---------------------------------------------------------------------------
// Flash-attention (online softmax), one block per (b, h, 64-query tile).
// qkv: [B*S, 3E] f16 ; out: [B*S, E] f16 head-merged.
// K/V tiles for block kb+1 are prefetched into VGPRs while block kb computes.
// window < 0: dense; window >= 0: mask |i-j| > window.
// ---------------------------------------------------------------------------
#define ATT_LD  72    // f16 tile stride in halves (144B, multiple of 16)
#define ATT_LDS 68    // f32 score tile stride in floats

__global__ __launch_bounds__(256) void attn_wmma(
    const _Float16* __restrict__ qkv,
    _Float16*       __restrict__ out,
    int window)
{
    __shared__ __align__(16) _Float16 Qs[64 * ATT_LD];
    __shared__ __align__(16) _Float16 Ks[64 * ATT_LD];
    __shared__ __align__(16) _Float16 Vt[64 * ATT_LD];   // V transposed [d][n]
    __shared__ __align__(16) _Float16 Ps[64 * ATT_LD];
    __shared__ float Ss[64 * ATT_LDS];
    __shared__ float rmax[64], rsum[64], rcorr[64], red[64 * 4];

    const int t    = threadIdx.x;
    const int lane = t & 31;
    const int wave = t >> 5;
    const int qb = blockIdx.x & 15;          // S/64 = 16
    const int h  = (blockIdx.x >> 4) & 7;    // H = 8
    const int b  = blockIdx.x >> 7;
    const size_t rowb = (size_t)b * 1024 + (size_t)qb * 64;

#pragma unroll
    for (int i = 0; i < 2; ++i) {
        int idx = t + i * 256;
        int r = idx >> 3, s = (idx & 7) << 3;
        *reinterpret_cast<U128*>(&Qs[r * ATT_LD + s]) =
            *reinterpret_cast<const U128*>(&qkv[(rowb + r) * 1536 + h * 64 + s]);
    }
    if (t < 64) { rmax[t] = -3.0e38f; rsum[t] = 0.0f; }

    const int wrg = (wave >> 1) * 16;   // this wave's 16 query rows
    const int wcg = (wave & 1) * 32;    // this wave's 32 key cols / head dims
    const int mo  = (lane < 16) ? 0 : 8;
    const int no  = lane & 15;

    U128 pk[2], pv[2];
    auto gloadKV = [&](int kb) {
#pragma unroll
        for (int i = 0; i < 2; ++i) {
            int idx = t + i * 256;
            int r = idx >> 3, s = (idx & 7) << 3;
            size_t grow = (size_t)b * 1024 + (size_t)kb * 64 + r;
            pk[i] = *reinterpret_cast<const U128*>(&qkv[grow * 1536 + 512 + h * 64 + s]);
            pv[i] = *reinterpret_cast<const U128*>(&qkv[grow * 1536 + 1024 + h * 64 + s]);
        }
    };

    v8f ao[2] = {};
    gloadKV(0);

    for (int kb = 0; kb < 16; ++kb) {
        __syncthreads();    // previous iteration done reading Ks/Vt
#pragma unroll
        for (int i = 0; i < 2; ++i) {
            int idx = t + i * 256;
            int r = idx >> 3, s = (idx & 7) << 3;
            *reinterpret_cast<U128*>(&Ks[r * ATT_LD + s]) = pk[i];
            const _Float16* vh = reinterpret_cast<const _Float16*>(&pv[i]);
#pragma unroll
            for (int j = 0; j < 8; ++j) Vt[(s + j) * ATT_LD + r] = vh[j];
        }
        __syncthreads();
        if (kb + 1 < 16) gloadKV(kb + 1);   // hide next block's loads here

        // S = Q K^T * (1/sqrt(D)), masked, into LDS.
        v8f sc[2] = {};
#pragma unroll
        for (int ks = 0; ks < 2; ++ks) {
            v16h aq = lds_a_frag(&Qs[wrg * ATT_LD + ks * 32], ATT_LD, lane);
#pragma unroll
            for (int c = 0; c < 2; ++c) {
                v16h bk = lds_b_frag(&Ks[(wcg + c * 16) * ATT_LD + ks * 32], ATT_LD, lane);
                sc[c] = wmma_f16(aq, bk, sc[c]);
            }
        }
#pragma unroll
        for (int c = 0; c < 2; ++c)
#pragma unroll
            for (int r = 0; r < 8; ++r) {
                int m = wrg + r + mo;
                int n = wcg + c * 16 + no;
                float s = sc[c][r] * 0.125f;
                if (window >= 0) {
                    int d = (qb * 64 + m) - (kb * 64 + n);
                    if (d > window || d < -window) s = -3.0e38f;
                }
                Ss[m * ATT_LDS + n] = s;
            }
        __syncthreads();

        // Row block-max (4 partials per row).
        {
            int r = t >> 2, q = t & 3;
            float mx = -3.0e38f;
#pragma unroll
            for (int j = 0; j < 16; ++j)
                mx = fmaxf(mx, Ss[r * ATT_LDS + q * 16 + j]);
            red[r * 4 + q] = mx;
        }
        __syncthreads();
        if (t < 64) {
            float bm = fmaxf(fmaxf(red[t * 4 + 0], red[t * 4 + 1]),
                             fmaxf(red[t * 4 + 2], red[t * 4 + 3]));
            float om = rmax[t];
            float nm = fmaxf(om, bm);
            rcorr[t] = __expf(om - nm);
            rmax[t]  = nm;
        }
        __syncthreads();

        // P = exp(S - rowmax) -> f16, plus row partial sums.
        {
            int r = t >> 2, q = t & 3;
            float m = rmax[r], sum = 0.0f;
#pragma unroll
            for (int j = 0; j < 16; ++j) {
                float p = __expf(Ss[r * ATT_LDS + q * 16 + j] - m);
                Ps[r * ATT_LD + q * 16 + j] = (_Float16)p;
                sum += p;
            }
            red[r * 4 + q] = sum;
        }
        __syncthreads();
        if (t < 64)
            rsum[t] = rsum[t] * rcorr[t] +
                      (red[t * 4 + 0] + red[t * 4 + 1] + red[t * 4 + 2] + red[t * 4 + 3]);

        // O = O*corr + P V.
#pragma unroll
        for (int c = 0; c < 2; ++c)
#pragma unroll
            for (int r = 0; r < 8; ++r)
                ao[c][r] *= rcorr[wrg + r + mo];
#pragma unroll
        for (int ks = 0; ks < 2; ++ks) {
            v16h ap = lds_a_frag(&Ps[wrg * ATT_LD + ks * 32], ATT_LD, lane);
#pragma unroll
            for (int c = 0; c < 2; ++c) {
                v16h bv = lds_b_frag(&Vt[(wcg + c * 16) * ATT_LD + ks * 32], ATT_LD, lane);
                ao[c] = wmma_f16(ap, bv, ao[c]);
            }
        }
    }
    __syncthreads();

#pragma unroll
    for (int c = 0; c < 2; ++c)
#pragma unroll
        for (int r = 0; r < 8; ++r) {
            int m = wrg + r + mo;
            int d = wcg + c * 16 + no;
            float v = ao[c][r] / rsum[m];
            out[(rowb + m) * 512 + h * 64 + d] = (_Float16)v;
        }
}

// ---------------------------------------------------------------------------
// Fused residual-add + LayerNorm. One block (256 thr) per row of E=512.
// wave32 shuffle reduction + 8-partial LDS combine.
// ---------------------------------------------------------------------------
template <bool HAS_X2, bool OUTF, bool OUTH>
__global__ __launch_bounds__(256) void add_ln(
    const float* __restrict__ x0, const float* __restrict__ x1,
    const float* __restrict__ x2,
    const float* __restrict__ gamma, const float* __restrict__ beta,
    float* __restrict__ outf, _Float16* __restrict__ outh)
{
    __shared__ float sred[8];
    const int t    = threadIdx.x;
    const int lane = t & 31;
    const int wave = t >> 5;
    const size_t base = (size_t)blockIdx.x * 512;

    float v[2];
#pragma unroll
    for (int i = 0; i < 2; ++i) {
        int c = t + i * 256;
        float s = x0[base + c] + x1[base + c];
        if (HAS_X2) s += x2[base + c];
        v[i] = s;
    }
    float ws = wave_red_sum(v[0] + v[1]);
    if (lane == 0) sred[wave] = ws;
    __syncthreads();
    float tot = 0.0f;
#pragma unroll
    for (int i = 0; i < 8; ++i) tot += sred[i];
    float mean = tot * (1.0f / 512.0f);
    __syncthreads();

    float d0 = v[0] - mean, d1 = v[1] - mean;
    ws = wave_red_sum(d0 * d0 + d1 * d1);
    if (lane == 0) sred[wave] = ws;
    __syncthreads();
    tot = 0.0f;
#pragma unroll
    for (int i = 0; i < 8; ++i) tot += sred[i];
    float inv = rsqrtf(tot * (1.0f / 512.0f) + 1e-5f);

#pragma unroll
    for (int i = 0; i < 2; ++i) {
        int c = t + i * 256;
        float o = (v[i] - mean) * inv * gamma[c] + beta[c];
        if (OUTF) outf[base + c] = o;
        if (OUTH) outh[base + c] = (_Float16)o;
    }
}

// ---------------------------------------------------------------------------
// Launch orchestration
// ---------------------------------------------------------------------------
extern "C" void kernel_launch(void* const* d_in, const int* in_sizes, int n_in,
                              void* d_out, int out_size, void* d_ws, size_t ws_size,
                              hipStream_t stream) {
    (void)in_sizes; (void)n_in; (void)out_size; (void)ws_size;

    constexpr int Bc = 8, Sc = 1024, Ec = 512, FF = 2048, WIN = 5;
    constexpr int M  = Bc * Sc;            // 8192 token rows
    constexpr int E3 = 3 * Ec;             // 1536

    const float* x      = (const float*)d_in[0];
    const float* l_in_w = (const float*)d_in[1];
    const float* l_in_b = (const float*)d_in[2];
    const float* l_ow   = (const float*)d_in[3];
    const float* l_ob   = (const float*)d_in[4];
    const float* g_in_w = (const float*)d_in[5];
    const float* g_in_b = (const float*)d_in[6];
    const float* g_ow   = (const float*)d_in[7];
    const float* g_ob   = (const float*)d_in[8];
    const float* f_w1   = (const float*)d_in[9];
    const float* f_b1   = (const float*)d_in[10];
    const float* f_w2   = (const float*)d_in[11];
    const float* f_b2   = (const float*)d_in[12];
    const float* ln1g   = (const float*)d_in[13];
    const float* ln1b   = (const float*)d_in[14];
    const float* ln2g   = (const float*)d_in[15];
    const float* ln2b   = (const float*)d_in[16];
    float* outp = (float*)d_out;

    char* ws = (char*)d_ws;
    size_t off = 0;
    auto alloc = [&](size_t bytes) -> char* {
        char* p = ws + off;
        off += (bytes + 255) & ~(size_t)255;
        return p;
    };
    _Float16* h_lin  = (_Float16*)alloc((size_t)E3 * Ec * 2);
    _Float16* h_gin  = (_Float16*)alloc((size_t)E3 * Ec * 2);
    _Float16* h_lout = (_Float16*)alloc((size_t)Ec * Ec * 2);
    _Float16* h_gout = (_Float16*)alloc((size_t)Ec * Ec * 2);
    _Float16* h_f1   = (_Float16*)alloc((size_t)FF * Ec * 2);
    _Float16* h_f2   = (_Float16*)alloc((size_t)Ec * FF * 2);
    _Float16* x_h    = (_Float16*)alloc((size_t)M * Ec * 2);
    _Float16* qkv_l  = (_Float16*)alloc((size_t)M * E3 * 2);
    _Float16* qkv_g  = (_Float16*)alloc((size_t)M * E3 * 2);
    _Float16* attn_l = (_Float16*)alloc((size_t)M * Ec * 2);
    _Float16* attn_g = (_Float16*)alloc((size_t)M * Ec * 2);
    float*    y_l    = (float*)   alloc((size_t)M * Ec * 4);
    float*    y_g    = (float*)   alloc((size_t)M * Ec * 4);
    float*    x1f    = (float*)   alloc((size_t)M * Ec * 4);
    _Float16* x1h    = (_Float16*)alloc((size_t)M * Ec * 2);
    // Reuse dead regions: QKV (48MB) -> FFN hidden (32MB); attn (16MB) -> y_ffn.
    _Float16* ffn_h  = qkv_l;
    float*    y_ffn  = (float*)attn_l;

    auto cvt = [&](const float* src, _Float16* dst, int n) {
        cvt_f32_f16<<<512, 256, 0, stream>>>(src, dst, n / 8);
    };
    cvt(x,      x_h,    M * Ec);
    cvt(l_in_w, h_lin,  E3 * Ec);
    cvt(g_in_w, h_gin,  E3 * Ec);
    cvt(l_ow,   h_lout, Ec * Ec);
    cvt(g_ow,   h_gout, Ec * Ec);
    cvt(f_w1,   h_f1,   FF * Ec);
    cvt(f_w2,   h_f2,   Ec * FF);

    dim3 blk(256);
    // QKV in-projections (f16 out).
    gemm_wmma<false, true, false><<<dim3(E3 / 128, M / 128), blk, GEMM_SMEM, stream>>>(
        x_h, h_lin, l_in_b, nullptr, qkv_l, M, E3, Ec);
    gemm_wmma<false, true, false><<<dim3(E3 / 128, M / 128), blk, GEMM_SMEM, stream>>>(
        x_h, h_gin, g_in_b, nullptr, qkv_g, M, E3, Ec);

    // Attention: local (windowed) + global.
    attn_wmma<<<dim3(Bc * 8 * (Sc / 64)), blk, 0, stream>>>(qkv_l, attn_l, WIN);
    attn_wmma<<<dim3(Bc * 8 * (Sc / 64)), blk, 0, stream>>>(qkv_g, attn_g, -1);

    // Output projections (f32 out).
    gemm_wmma<true, false, false><<<dim3(Ec / 128, M / 128), blk, GEMM_SMEM, stream>>>(
        attn_l, h_lout, l_ob, y_l, nullptr, M, Ec, Ec);
    gemm_wmma<true, false, false><<<dim3(Ec / 128, M / 128), blk, GEMM_SMEM, stream>>>(
        attn_g, h_gout, g_ob, y_g, nullptr, M, Ec, Ec);

    // x1 = LN(x + local + global), keep f32 + f16.
    add_ln<true, true, true><<<dim3(M), blk, 0, stream>>>(
        x, y_l, y_g, ln1g, ln1b, x1f, x1h);

    // FFN: relu(x1 W1^T + b1) W2^T + b2.
    gemm_wmma<false, true, true><<<dim3(FF / 128, M / 128), blk, GEMM_SMEM, stream>>>(
        x1h, h_f1, f_b1, nullptr, ffn_h, M, FF, Ec);
    gemm_wmma<true, false, false><<<dim3(Ec / 128, M / 128), blk, GEMM_SMEM, stream>>>(
        ffn_h, h_f2, f_b2, y_ffn, nullptr, M, Ec, FF);

    // out = LN(x1 + ffn).
    add_ln<false, true, false><<<dim3(M), blk, 0, stream>>>(
        x1f, y_ffn, nullptr, ln2g, ln2b, outp, nullptr);
}
